// MultiHeadDifferentialAttention_71184787964052
// MI455X (gfx1250) — compile-verified, matrix operand order checked by reading, not run on hardware
//
#include <hip/hip_runtime.h>
#include <hip/hip_bf16.h>
#include <math.h>

typedef __bf16 bfloat;
typedef __attribute__((ext_vector_type(4)))  __bf16 bf16x4;
typedef __attribute__((ext_vector_type(8)))  __bf16 bf16x8;
typedef __attribute__((ext_vector_type(16))) __bf16 v16bf;
typedef __attribute__((ext_vector_type(8)))  float  v8f;
typedef __attribute__((ext_vector_type(4)))  int    v4i;

#define SEQ 1024
#define NHEADS 16
#define DHEAD 64
#define LAMBDA_INIT 0.8f

// ---------------------------------------------------------------------------
// helpers
// ---------------------------------------------------------------------------
__device__ __forceinline__ v8f vzero8() {
  v8f v;
#pragma unroll
  for (int i = 0; i < 8; ++i) v[i] = 0.0f;
  return v;
}

// Async global->LDS copy of one 16B chunk (GLOBAL_LOAD_ASYNC_TO_LDS_B128,
// tracked by ASYNCcnt; bypasses VGPRs entirely).
// Builtin signature (from hipcc diagnostic): (int4 AS1*, int4 AS3*, imm, imm).
__device__ __forceinline__ void async_b128(bfloat* lds, const bfloat* gmem) {
  __builtin_amdgcn_global_load_async_to_lds_b128(
      (__attribute__((address_space(1))) v4i*)(gmem),
      (__attribute__((address_space(3))) v4i*)(lds), 0, 0);
}

#define S_WAIT_ASYNCCNT0() asm volatile("s_wait_asynccnt 0x0" ::: "memory")

// Load one 16x32 bf16 A/B fragment from a row-major tile in LDS.
// Per CDNA5 ISA 7.12.2: lane L holds row M=L%16, K in {k0..k0+7, k0+16..k0+23}
// with k0 = 8*(L>=16). Two ds_load_b128 per lane.
__device__ __forceinline__ v16bf frag_ld(const bfloat* __restrict__ base,
                                         int ldh, int row, int k0) {
  const bfloat* p = base + row * ldh + k0;
  bf16x8 lo = *(const bf16x8*)(p);
  bf16x8 hi = *(const bf16x8*)(p + 16);
  v16bf r;
#pragma unroll
  for (int i = 0; i < 8; ++i) { r[i] = lo[i]; r[i + 8] = hi[i]; }
  return r;
}

__device__ __forceinline__ v8f wmma_bf16(v16bf a, v16bf b, v8f c) {
  return __builtin_amdgcn_wmma_f32_16x16x32_bf16(false, a, false, b,
                                                 (short)0, c, false, false);
}

// Reductions across the 16-lane halves of a wave32 (matches WMMA C layout:
// lanes 0-15 carry row r, lanes 16-31 carry row r+8).
__device__ __forceinline__ float redmax16(float v) {
  v = fmaxf(v, __shfl_xor(v, 1));
  v = fmaxf(v, __shfl_xor(v, 2));
  v = fmaxf(v, __shfl_xor(v, 4));
  v = fmaxf(v, __shfl_xor(v, 8));
  return v;
}
__device__ __forceinline__ float redsum16(float v) {
  v += __shfl_xor(v, 1);
  v += __shfl_xor(v, 2);
  v += __shfl_xor(v, 4);
  v += __shfl_xor(v, 8);
  return v;
}

// ---------------------------------------------------------------------------
// f32 -> bf16 conversion (float4 -> 4x bf16 per thread)
// ---------------------------------------------------------------------------
__global__ __launch_bounds__(256) void cvt_f32_bf16_kernel(
    const float* __restrict__ src, bfloat* __restrict__ dst, int n4) {
  int i = blockIdx.x * 256 + threadIdx.x;
  if (i >= n4) return;
  float4 v = ((const float4*)src)[i];
  bf16x4 o;
  o[0] = (bfloat)v.x; o[1] = (bfloat)v.y; o[2] = (bfloat)v.z; o[3] = (bfloat)v.w;
  ((bf16x4*)dst)[i] = o;
}

// f32[K][N] -> bf16 transposed [N][K] (weights only; tiny, runs once)
__global__ __launch_bounds__(256) void cvt_transpose_bf16_kernel(
    const float* __restrict__ src, bfloat* __restrict__ dst, int K, int N) {
  int idx = blockIdx.x * 256 + threadIdx.x;
  if (idx >= N * K) return;
  int n = idx / K, k = idx - n * K;
  dst[idx] = (bfloat)src[(size_t)k * N + n];
}

// ---------------------------------------------------------------------------
// per-head lambda = exp(lq1.lk1) - exp(lq2.lk2) + 0.8
// ---------------------------------------------------------------------------
__global__ void lambda_kernel(const float* __restrict__ lq1,
                              const float* __restrict__ lk1,
                              const float* __restrict__ lq2,
                              const float* __restrict__ lk2,
                              float* __restrict__ lam) {
  int h = threadIdx.x;
  if (h >= NHEADS) return;
  float s1 = 0.f, s2 = 0.f;
  for (int e = 0; e < DHEAD; ++e) {
    s1 += lq1[h * DHEAD + e] * lk1[h * DHEAD + e];
    s2 += lq2[h * DHEAD + e] * lk2[h * DHEAD + e];
  }
  lam[h] = __expf(s1) - __expf(s2) + LAMBDA_INIT;
}

// ---------------------------------------------------------------------------
// Tiled bf16 WMMA GEMM: C = A[M,K] * Bt[N,K]^T  (both operands K-contiguous)
// Block tile 128x128, K-step 32, double-buffered async global->LDS copies.
// 8 waves, each computing a 32x64 sub-tile as 2x4 WMMA accumulators.
// OUT_MODE: 0 = f32 C[M,N]; 1 = bf16 C[M,N]; 2 = bf16 transposed C_T[N,M]
// ---------------------------------------------------------------------------
template <int OUT_MODE>
__global__ __launch_bounds__(256) void gemm_bf16_kernel(
    const bfloat* __restrict__ A, const bfloat* __restrict__ Bt,
    void* __restrict__ C, int M, int N, int K) {
  __shared__ __align__(16) bfloat As[2][128 * 32];  // [m][k]
  __shared__ __align__(16) bfloat Bs[2][128 * 32];  // [n][k]

  const int tid  = threadIdx.x;
  const int lane = tid & 31;
  const int wave = tid >> 5;
  const int lm   = lane & 15;
  const int lh   = lane >> 4;
  const int bm   = blockIdx.y * 128;
  const int bn   = blockIdx.x * 128;
  const int wm   = (wave >> 1) * 32;
  const int wn   = (wave & 1) * 64;
  const int KT   = K >> 5;

  auto issue = [&](int kt, int buf) {
#pragma unroll
    for (int it = 0; it < 2; ++it) {
      int c = tid + it * 256;
      int r = c >> 2, kc = (c & 3) * 8;
      async_b128(&As[buf][r * 32 + kc],
                 &A[(size_t)(bm + r) * K + kt * 32 + kc]);
      async_b128(&Bs[buf][r * 32 + kc],
                 &Bt[(size_t)(bn + r) * K + kt * 32 + kc]);
    }
  };

  v8f acc[2][4];
#pragma unroll
  for (int i = 0; i < 2; ++i)
#pragma unroll
    for (int j = 0; j < 4; ++j) acc[i][j] = vzero8();

  issue(0, 0);
  for (int kt = 0; kt < KT; ++kt) {
    const int cur = kt & 1;
    S_WAIT_ASYNCCNT0();   // my chunks of tile kt have landed in LDS
    __syncthreads();      // everyone's landed; everyone done reading buf cur^1
    if (kt + 1 < KT) issue(kt + 1, cur ^ 1);  // prefetch overlaps compute

    v16bf af[2], bfr[4];
#pragma unroll
    for (int mi = 0; mi < 2; ++mi)
      af[mi] = frag_ld(As[cur], 32, wm + mi * 16 + lm, 8 * lh);
#pragma unroll
    for (int ni = 0; ni < 4; ++ni)
      bfr[ni] = frag_ld(Bs[cur], 32, wn + ni * 16 + lm, 8 * lh);
#pragma unroll
    for (int mi = 0; mi < 2; ++mi)
#pragma unroll
      for (int ni = 0; ni < 4; ++ni)
        acc[mi][ni] = wmma_bf16(af[mi], bfr[ni], acc[mi][ni]);
  }

  // C layout: VGPR r -> (row r, lanes 0-15) and (row r+8, lanes 16-31)
#pragma unroll
  for (int mi = 0; mi < 2; ++mi)
#pragma unroll
    for (int ni = 0; ni < 4; ++ni) {
      if (OUT_MODE == 2) {
        // transposed store: 8 consecutive rows per lane -> one b128 store
        bf16x8 v;
#pragma unroll
        for (int r = 0; r < 8; ++r) v[r] = (bfloat)acc[mi][ni][r];
        int col  = bn + wn + ni * 16 + lm;
        int row0 = bm + wm + mi * 16 + 8 * lh;
        *(bf16x8*)&((bfloat*)C)[(size_t)col * M + row0] = v;
      } else {
#pragma unroll
        for (int r = 0; r < 8; ++r) {
          int row = bm + wm + mi * 16 + r + 8 * lh;
          int col = bn + wn + ni * 16 + lm;
          if (OUT_MODE == 1)
            ((bfloat*)C)[(size_t)row * N + col] = (bfloat)acc[mi][ni][r];
          else
            ((float*)C)[(size_t)row * N + col] = acc[mi][ni][r];
        }
      }
    }
}

// ---------------------------------------------------------------------------
// Differential flash attention.
// Grid: B*NHEADS*(SEQ/64) blocks; 128 threads (4 waves, 16 q-rows per wave).
// K/V tiles stream through double-buffered async global->LDS copies; two
// online softmaxes (s1, s2) share one non-temporal mask read; O1/O2
// accumulated with WMMA; epilogue: rmsnorm(O1/l1 - lam*O2/l2)*scale*0.2.
// ---------------------------------------------------------------------------
__global__ __launch_bounds__(128) void diff_attn_kernel(
    const bfloat* __restrict__ Qp, const bfloat* __restrict__ Kp,
    const bfloat* __restrict__ VPt, const float* __restrict__ mask,
    const float* __restrict__ lam_arr, const float* __restrict__ rms_scale,
    bfloat* __restrict__ AT) {
  __shared__ __align__(16) bfloat Qs1[64 * 64];
  __shared__ __align__(16) bfloat Qs2[64 * 64];
  __shared__ __align__(16) bfloat Ks1[2][64 * 64];  // [key][e]
  __shared__ __align__(16) bfloat Ks2[2][64 * 64];  // [key][e]
  __shared__ __align__(16) bfloat Vt[2][64 * 64];   // [e][key] (pre-transposed)
  __shared__ __align__(16) bfloat Ps[4 * 16 * 64];  // per-wave P staging

  const int tid  = threadIdx.x;
  const int lane = tid & 31;
  const int wave = tid >> 5;
  const int lm   = lane & 15;
  const int lh   = lane >> 4;

  const int blk = blockIdx.x;
  const int qt  = blk & 15;
  const int h   = (blk >> 4) & 15;
  const int b   = blk >> 8;

  const float lam     = lam_arr[h];
  const float scaling = 0.125f;  // 1/sqrt(64)

  const size_t vtrow0 = (size_t)(h * 64) * 4096 + (size_t)(b * SEQ);

  auto issueKV = [&](int kt, int buf) {
    const size_t kg = ((size_t)(b * SEQ + kt * 64)) * 2048 + (size_t)h * 128;
    const size_t vg = vtrow0 + (size_t)(kt * 64);
#pragma unroll
    for (int it = 0; it < 4; ++it) {
      int c = tid + it * 128;
      int r = c >> 3, cc = (c & 7) * 8;
      async_b128(&Ks1[buf][r * 64 + cc], &Kp[kg + (size_t)r * 2048 + cc]);
      async_b128(&Ks2[buf][r * 64 + cc], &Kp[kg + (size_t)r * 2048 + 64 + cc]);
      async_b128(&Vt[buf][r * 64 + cc], &VPt[vg + (size_t)r * 4096 + cc]);
    }
  };

  issueKV(0, 0);  // prefetch first K/V tile while Q is staged

  // Q rows for this tile: Qproj[(b*S + qt*64 + r), h*128 + (0..127)]
  const size_t qgbase =
      ((size_t)(b * SEQ + qt * 64)) * 2048 + (size_t)h * 128;
#pragma unroll
  for (int it = 0; it < 4; ++it) {
    int c = tid + it * 128;
    int r = c >> 3, cc = (c & 7) * 8;
    *(bf16x8*)&Qs1[r * 64 + cc] =
        *(const bf16x8*)&Qp[qgbase + (size_t)r * 2048 + cc];
    *(bf16x8*)&Qs2[r * 64 + cc] =
        *(const bf16x8*)&Qp[qgbase + (size_t)r * 2048 + 64 + cc];
  }
  __syncthreads();

  v16bf qa1[2], qa2[2];
#pragma unroll
  for (int ch = 0; ch < 2; ++ch) {
    qa1[ch] = frag_ld(Qs1, 64, wave * 16 + lm, ch * 32 + 8 * lh);
    qa2[ch] = frag_ld(Qs2, 64, wave * 16 + lm, ch * 32 + 8 * lh);
  }

  v8f O1[4], O2[4];
  float m1[8], l1[8], m2[8], l2[8];
#pragma unroll
  for (int f = 0; f < 4; ++f) { O1[f] = vzero8(); O2[f] = vzero8(); }
#pragma unroll
  for (int r = 0; r < 8; ++r) {
    m1[r] = -1e30f; l1[r] = 0.f; m2[r] = -1e30f; l2[r] = 0.f;
  }

  const float* mrow = mask + (size_t)(b * NHEADS + h) * (size_t)SEQ * SEQ;
  const int qglob   = qt * 64 + wave * 16;
  bfloat* pw        = Ps + wave * 16 * 64;

  for (int kt = 0; kt < 16; ++kt) {
    const int cur = kt & 1;
    S_WAIT_ASYNCCNT0();  // my chunks of K/V tile kt landed
    __syncthreads();     // all landed; all done reading buffer cur^1
    if (kt + 1 < 16) issueKV(kt + 1, cur ^ 1);

    // mask tile: read once (non-temporal; 256MB streamed exactly once),
    // shared by both score paths
    v8f mk[4];
#pragma unroll
    for (int nt = 0; nt < 4; ++nt)
#pragma unroll
      for (int r = 0; r < 8; ++r)
        mk[nt][r] = __builtin_nontemporal_load(
            &mrow[(size_t)(qglob + r + 8 * lh) * SEQ + kt * 64 + nt * 16 + lm]);

    auto path = [&](const bfloat* Ks, const v16bf* qa, float* mr, float* lr,
                    v8f* O) {
      v8f s[4];
#pragma unroll
      for (int nt = 0; nt < 4; ++nt) {
        v16bf b0 = frag_ld(Ks, 64, nt * 16 + lm, 8 * lh);
        v16bf b1 = frag_ld(Ks, 64, nt * 16 + lm, 32 + 8 * lh);
        v8f c = vzero8();
        c = wmma_bf16(qa[0], b0, c);
        c = wmma_bf16(qa[1], b1, c);
#pragma unroll
        for (int r = 0; r < 8; ++r) s[nt][r] = c[r] * scaling + mk[nt][r];
      }
      // online softmax update
      float corr[8];
#pragma unroll
      for (int r = 0; r < 8; ++r) {
        float tm = fmaxf(fmaxf(s[0][r], s[1][r]), fmaxf(s[2][r], s[3][r]));
        tm = redmax16(tm);
        float mn = fmaxf(mr[r], tm);
        corr[r] = __expf(mr[r] - mn);
        mr[r] = mn;
        lr[r] *= corr[r];
      }
#pragma unroll
      for (int f = 0; f < 4; ++f)
#pragma unroll
        for (int r = 0; r < 8; ++r) O[f][r] *= corr[r];
      v8f p[4];
#pragma unroll
      for (int nt = 0; nt < 4; ++nt)
#pragma unroll
        for (int r = 0; r < 8; ++r) p[nt][r] = __expf(s[nt][r] - mr[r]);
#pragma unroll
      for (int r = 0; r < 8; ++r) {
        float t = p[0][r] + p[1][r] + p[2][r] + p[3][r];
        lr[r] += redsum16(t);
      }
      // stage P via per-wave LDS region: C-layout -> A-layout
#pragma unroll
      for (int nt = 0; nt < 4; ++nt)
#pragma unroll
        for (int r = 0; r < 8; ++r)
          pw[(r + 8 * lh) * 64 + nt * 16 + lm] = (bfloat)p[nt][r];
      v16bf pa0 = frag_ld(pw, 64, lm, 8 * lh);
      v16bf pa1 = frag_ld(pw, 64, lm, 32 + 8 * lh);
#pragma unroll
      for (int f = 0; f < 4; ++f) {
        v16bf bv0 = frag_ld(Vt[cur], 64, f * 16 + lm, 8 * lh);
        v16bf bv1 = frag_ld(Vt[cur], 64, f * 16 + lm, 32 + 8 * lh);
        O[f] = wmma_bf16(pa0, bv0, O[f]);
        O[f] = wmma_bf16(pa1, bv1, O[f]);
      }
    };
    path(Ks1[cur], qa1, m1, l1, O1);
    path(Ks2[cur], qa2, m2, l2, O2);
  }

  // combine: out = O1/l1 - lam*O2/l2, then RMS norm over the 64-wide head dim
  v8f o[4];
#pragma unroll
  for (int r = 0; r < 8; ++r) {
    float i1 = 1.0f / l1[r];
    float i2 = lam / l2[r];
#pragma unroll
    for (int f = 0; f < 4; ++f) o[f][r] = O1[f][r] * i1 - O2[f][r] * i2;
  }
  float sc[4];
#pragma unroll
  for (int f = 0; f < 4; ++f) sc[f] = rms_scale[f * 16 + lm];

  const size_t obase = ((size_t)(b * SEQ + qglob)) * 1024 + (size_t)h * 64;
#pragma unroll
  for (int r = 0; r < 8; ++r) {
    float t = 0.f;
#pragma unroll
    for (int f = 0; f < 4; ++f) t += o[f][r] * o[f][r];
    t = redsum16(t);
    float rinv = rsqrtf(t * (1.0f / 64.f) + 1e-5f);
#pragma unroll
    for (int f = 0; f < 4; ++f) {
      float v = o[f][r] * rinv * sc[f] * (1.0f - LAMBDA_INIT);
      AT[obase + (size_t)(r + 8 * lh) * 1024 + f * 16 + lm] = (bfloat)v;
    }
  }
}

// ---------------------------------------------------------------------------
// launch
// ---------------------------------------------------------------------------
extern "C" void kernel_launch(void* const* d_in, const int* in_sizes, int n_in,
                              void* d_out, int out_size, void* d_ws,
                              size_t ws_size, hipStream_t stream) {
  (void)in_sizes; (void)n_in; (void)out_size; (void)ws_size;

  const float* query = (const float*)d_in[0];
  const float* key   = (const float*)d_in[1];
  const float* value = (const float*)d_in[2];
  const float* mask  = (const float*)d_in[3];
  const float* Wq    = (const float*)d_in[4];
  const float* Wk    = (const float*)d_in[5];
  const float* Wv    = (const float*)d_in[6];
  const float* Wo    = (const float*)d_in[7];
  const float* lq1   = (const float*)d_in[8];
  const float* lk1   = (const float*)d_in[9];
  const float* lq2   = (const float*)d_in[10];
  const float* lk2   = (const float*)d_in[11];
  const float* rscl  = (const float*)d_in[12];

  char* ws = (char*)d_ws;
  // workspace layout (bytes)
  bfloat* Xq   = (bfloat*)(ws + 0);          //  8 MiB (4096x1024 bf16)
  bfloat* Xk   = (bfloat*)(ws + 8388608);    //  8 MiB
  bfloat* Xv   = (bfloat*)(ws + 16777216);   //  8 MiB
  bfloat* WqBt = (bfloat*)(ws + 25165824);   //  4 MiB (2048x1024, transposed)
  bfloat* WkBt = (bfloat*)(ws + 29360128);   //  4 MiB
  bfloat* WvBt = (bfloat*)(ws + 33554432);   //  2 MiB (1024x1024, transposed)
  bfloat* WoBt = (bfloat*)(ws + 35651584);   //  2 MiB
  bfloat* QP   = (bfloat*)(ws + 37748736);   // 16 MiB (4096x2048)
  bfloat* KP   = (bfloat*)(ws + 54525952);   // 16 MiB
  bfloat* VPt  = (bfloat*)(ws + 71303168);   //  8 MiB (1024x4096, transposed)
  bfloat* AT   = (bfloat*)(ws + 79691776);   //  8 MiB (4096x1024)
  float*  LAM  = (float*)(ws + 88080384);    //  64 B

  // f32 -> bf16 conversions (activations row-major, weights transposed)
  cvt_f32_bf16_kernel<<<4096, 256, 0, stream>>>(query, Xq, 1048576);
  cvt_f32_bf16_kernel<<<4096, 256, 0, stream>>>(key,   Xk, 1048576);
  cvt_f32_bf16_kernel<<<4096, 256, 0, stream>>>(value, Xv, 1048576);
  cvt_transpose_bf16_kernel<<<8192, 256, 0, stream>>>(Wq, WqBt, 1024, 2048);
  cvt_transpose_bf16_kernel<<<8192, 256, 0, stream>>>(Wk, WkBt, 1024, 2048);
  cvt_transpose_bf16_kernel<<<4096, 256, 0, stream>>>(Wv, WvBt, 1024, 1024);
  cvt_transpose_bf16_kernel<<<4096, 256, 0, stream>>>(Wo, WoBt, 1024, 1024);

  lambda_kernel<<<1, 16, 0, stream>>>(lq1, lk1, lq2, lk2, LAM);

  // projections (bf16 WMMA, f32 accumulate)
  gemm_bf16_kernel<1><<<dim3(16, 32), 256, 0, stream>>>(
      Xq, WqBt, QP, 4096, 2048, 1024);
  gemm_bf16_kernel<1><<<dim3(16, 32), 256, 0, stream>>>(
      Xk, WkBt, KP, 4096, 2048, 1024);
  // V projection stored transposed so attention V-tiles are async-copyable
  gemm_bf16_kernel<2><<<dim3(8, 32), 256, 0, stream>>>(
      Xv, WvBt, VPt, 4096, 1024, 1024);

  // differential flash attention: B*H*(S/64) = 1024 blocks
  diff_attn_kernel<<<1024, 128, 0, stream>>>(QP, KP, VPt, mask, LAM, rscl, AT);

  // output projection -> f32 d_out
  gemm_bf16_kernel<0><<<dim3(8, 32), 256, 0, stream>>>(
      AT, WoBt, d_out, 4096, 1024, 1024);
}
